// CausalSelfAttention_51032801411712
// MI455X (gfx1250) — matrix operand order repaired and checked by reading, not compiled, not run on hardware
//
#include <hip/hip_runtime.h>
#include <hip/hip_bf16.h>

typedef __attribute__((ext_vector_type(16))) _Float16 v16h;
typedef __attribute__((ext_vector_type(8)))  _Float16 v8h;
typedef __attribute__((ext_vector_type(4)))  _Float16 v4h;
typedef __attribute__((ext_vector_type(8)))  float    v8f;

// ---------------------------------------------------------------------------
// CDNA5 WMMA: D = A(16x32 f16) x B(32x16 f16) + C(16x16 f32)
// ---------------------------------------------------------------------------
static __device__ __forceinline__ v8f wmma_f16(v16h a, v16h b, v8f c) {
    return __builtin_amdgcn_wmma_f32_16x16x32_f16(
        false, a, false, b, (short)0, c, false, false);
}

// Fragment index maps (contiguous k order used by this file, not the ISA table order):
// A 16x32 (MxK): lane%16 = M; element i -> K = i + (lane>=16?16:0)
static __device__ __forceinline__ int ka_idx(int i, int lane) {
    return i + ((lane & 16) ? 16 : 0);
}
// B 32x16 (KxN): lane%16 = N; element i -> K = i + (lane>=16?16:0)
// C/D 16x16 f32: lane%16 = N; vgpr r -> M = r + (lane>=16?8:0)

// ---------------------------------------------------------------------------
// CDNA5 async global->LDS copy (ASYNCcnt path), 16 bytes per enabled lane.
// ---------------------------------------------------------------------------
static __device__ __forceinline__ void async_ld_b128(void* lds_ptr, const void* gptr) {
    unsigned lds_off = (unsigned)(size_t)lds_ptr;   // low 32 bits of flat addr == LDS offset
    asm volatile("global_load_async_to_lds_b128 %0, %1, off"
                 :: "v"(lds_off), "v"(gptr)
                 : "memory");
}
template <int N>
static __device__ __forceinline__ void async_wait() {
    asm volatile("s_wait_asynccnt %0" :: "n"(N) : "memory");
}

// ---------------------------------------------------------------------------
// Half-wave (16-lane) XOR reductions via ds_swizzle (1 instr per step).
// Group-of-32 encoding: and=0x1f, or=0, xor in bits [14:10].
// ---------------------------------------------------------------------------
template <int PAT>
static __device__ __forceinline__ float swz(float x) {
    return __int_as_float(__builtin_amdgcn_ds_swizzle(__float_as_int(x), PAT));
}
static __device__ __forceinline__ float rmax16(float x) {
    x = fmaxf(x, swz<0x041f>(x));
    x = fmaxf(x, swz<0x081f>(x));
    x = fmaxf(x, swz<0x101f>(x));
    x = fmaxf(x, swz<0x201f>(x));
    return x;
}
static __device__ __forceinline__ float rsum16(float x) {
    x += swz<0x041f>(x);
    x += swz<0x081f>(x);
    x += swz<0x101f>(x);
    x += swz<0x201f>(x);
    return x;
}

// ---------------------------------------------------------------------------
// f32 -> f16 bulk convert (vectorized x4)
// ---------------------------------------------------------------------------
__global__ __launch_bounds__(256) void cvt16(const float* __restrict__ in,
                                             _Float16* __restrict__ out, int n4) {
    const int i = blockIdx.x * 256 + threadIdx.x;
    if (i >= n4) return;
    const float4 f = reinterpret_cast<const float4*>(in)[i];
    v4h h; h[0] = (_Float16)f.x; h[1] = (_Float16)f.y;
           h[2] = (_Float16)f.z; h[3] = (_Float16)f.w;
    reinterpret_cast<v4h*>(out)[i] = h;
}

// ---------------------------------------------------------------------------
// LDS-tiled WMMA GEMM, double-buffered async pipeline.
// out[m][n] = sum_k A[m][k]*W[n][k] + bias[n]
// Block = 16x128 output tile (8 waves x one 16x16 tile each), k-chunk = 64.
// M%16==0, N%128==0, K%64==0.
// ---------------------------------------------------------------------------
template <typename TOUT>
__global__ __launch_bounds__(256) void gemm16lds(const _Float16* __restrict__ A,
                                                 const _Float16* __restrict__ W,
                                                 const float* __restrict__ bias,
                                                 TOUT* __restrict__ out,
                                                 int M, int N, int K) {
    __shared__ _Float16 a_tile[2][16][72];    // 16 x 64 (+8 pad, 144B stride)
    __shared__ _Float16 w_tile[2][128][72];   // 128 x 64

    const int t    = threadIdx.x;
    const int lane = t & 31;
    const int w    = t >> 5;
    const int nblks = N >> 7;                  // N/128
    const int mt = blockIdx.x / nblks;
    const int nb = blockIdx.x % nblks;
    const size_t M0  = (size_t)mt << 4;
    const size_t N0b = (size_t)nb << 7;
    const int am   = lane & 15;
    const int hi8  = (lane & 16) ? 8 : 0;
    const int hi16 = (lane & 16) ? 16 : 0;
    const size_t N0 = N0b + 16 * w;

    const int srow = t >> 3;                   // 0..31
    const int scol = (t & 7) * 8;              // 0..56

    auto stage = [&](int buf, int k0) {
        if (t < 128)                           // waves 0..3: A tile 16x64
            async_ld_b128(&a_tile[buf][srow & 15][scol],
                          A + (M0 + (srow & 15)) * (size_t)K + k0 + scol);
#pragma unroll
        for (int s = 0; s < 4; ++s)            // all waves: W tile 128x64
            async_ld_b128(&w_tile[buf][srow + 32 * s][scol],
                          W + (N0b + srow + 32 * s) * (size_t)K + k0 + scol);
    };

    v8f acc = {};
    stage(0, 0);
    const int niter = K >> 6;                  // K/64
    for (int i = 0; i < niter; ++i) {
        async_wait<0>();
        __syncthreads();                       // stage(i) visible; compute(i-1) done
        if (i + 1 < niter) stage((i + 1) & 1, (i + 1) * 64);   // overlaps compute(i)

        const int buf = i & 1;
        v16h a0, a1, w0, w1;
#pragma unroll
        for (int j = 0; j < 16; ++j) {
            const int kk = ka_idx(j, lane);
            a0[j] = a_tile[buf][am][kk];
            a1[j] = a_tile[buf][am][32 + kk];
            w0[j] = w_tile[buf][16 * w + am][j + hi16];
            w1[j] = w_tile[buf][16 * w + am][32 + j + hi16];
        }
        acc = wmma_f16(a0, w0, acc);
        acc = wmma_f16(a1, w1, acc);
    }

    const float bi = bias[N0 + am];
#pragma unroll
    for (int r = 0; r < 8; ++r) {
        const size_t row = M0 + r + hi8;
        out[row * (size_t)N + N0 + am] = (TOUT)(acc[r] + bi);
    }
}

// ---------------------------------------------------------------------------
// Block-cooperative causal flash attention, double-buffered K/V staging,
// work-balanced stripe pairing.
// Each block handles TWO 128-row query stripes of one (b,h): stripe p and
// stripe (31-p), so every block runs exactly 132 key-iterations (no tail).
// Wave w owns rows [16w,16w+16) of the current stripe.
// K tile: async global->LDS. V tile: b128 load issued before compute,
// transpose-stored after compute (latency hidden behind 8 WMMAs).
// ---------------------------------------------------------------------------
__global__ __launch_bounds__(256) void flash16(const _Float16* __restrict__ qkv,
                                               _Float16* __restrict__ y) {
    constexpr int T = 4096, C3 = 1536, Cc = 512;
    constexpr int NSTRIPES = T / 128;          // 32

    __shared__ _Float16 k_tile[2][32][72];   // [key][d]  (144B stride)
    __shared__ _Float16 vt_tile[2][64][40];  // [d][key]  (80B stride)
    __shared__ _Float16 plds[8][16][40];     // per-wave P tile (80B stride)

    const int t    = threadIdx.x;
    const int lane = t & 31;
    const int w    = t >> 5;
    const int npairs = NSTRIPES / 2;           // 16
    const int bh   = blockIdx.x / npairs;
    const int pair = blockIdx.x % npairs;
    const int b = bh >> 3, h = bh & 7;

    const size_t base = (size_t)b * T * C3 + (size_t)h * 64;
    const _Float16* qp = qkv + base;
    const _Float16* kp = qkv + base + 512;
    const _Float16* vp = qkv + base + 1024;

    const int am   = lane & 15;
    const int hi8  = (lane & 16) ? 8 : 0;
    const int hi16 = (lane & 16) ? 16 : 0;

    const int skey = t >> 3;                 // 0..31
    const int sd   = (t & 7) * 8;            // 0..56

    for (int half = 0; half < 2; ++half) {
        const int stripe = (half == 0) ? pair : (NSTRIPES - 1 - pair);
        const int qbase0 = stripe * 128;
        const int qbase  = qbase0 + 16 * w;

        __syncthreads();                     // LDS tiles free before (re)staging

        // Q fragments: 16 rows x 64 dims, resident in registers
        v16h qf0, qf1;
        {
            const size_t qrow = (size_t)(qbase + am) * C3;
#pragma unroll
            for (int i = 0; i < 16; ++i) {
                const int kk = ka_idx(i, lane);
                qf0[i] = qp[qrow + kk];
                qf1[i] = qp[qrow + 32 + kk];
            }
        }

        v8f o0 = {}, o1 = {}, o2 = {}, o3 = {};
        float mrow[8], lrow[8];
#pragma unroll
        for (int r = 0; r < 8; ++r) { mrow[r] = -3.0e38f; lrow[r] = 0.0f; }

        const int niter = (qbase0 + 128) / 32;

        // prologue: stage key-block 0
        async_ld_b128(&k_tile[0][skey][sd], kp + (size_t)skey * C3 + sd);
        {
            const v8h vv0 = *reinterpret_cast<const v8h*>(vp + (size_t)skey * C3 + sd);
#pragma unroll
            for (int j = 0; j < 8; ++j) vt_tile[0][sd + j][skey] = vv0[j];
        }

        for (int i = 0; i < niter; ++i) {
            const int kb = i * 32;
            async_wait<0>();
            __syncthreads();                 // stage(i) visible; compute(i-1) done

            // issue next stage: async K, V load (consumed after compute)
            v8h vv;
            const bool have_next = (i + 1 < niter);
            if (have_next) {
                const int kn = kb + 32;
                async_ld_b128(&k_tile[(i + 1) & 1][skey][sd],
                              kp + (size_t)(kn + skey) * C3 + sd);
                vv = *reinterpret_cast<const v8h*>(vp + (size_t)(kn + skey) * C3 + sd);
            }

            if (kb <= qbase + 15) {          // causal: this wave still has work
                const int cb = i & 1;
                // ----- scores: S[16q x 32k] -----
                v8f s0 = {}, s1 = {};
                {
                    v16h kf00, kf01, kf10, kf11;
#pragma unroll
                    for (int i2 = 0; i2 < 16; ++i2) {
                        kf00[i2] = k_tile[cb][am][i2 + hi16];
                        kf01[i2] = k_tile[cb][am][32 + i2 + hi16];
                        kf10[i2] = k_tile[cb][16 + am][i2 + hi16];
                        kf11[i2] = k_tile[cb][16 + am][32 + i2 + hi16];
                    }
                    s0 = wmma_f16(qf0, kf00, s0);
                    s0 = wmma_f16(qf1, kf01, s0);
                    s1 = wmma_f16(qf0, kf10, s1);
                    s1 = wmma_f16(qf1, kf11, s1);
                }

                // ----- scale, causal mask, online softmax -----
#pragma unroll
                for (int r = 0; r < 8; ++r) {
                    const int row = qbase + r + hi8;
                    float v0 = s0[r] * 0.125f;
                    float v1 = s1[r] * 0.125f;
                    if (kb + am > row)      v0 = -3.0e38f;
                    if (kb + 16 + am > row) v1 = -3.0e38f;

                    const float mx   = rmax16(fmaxf(v0, v1));
                    const float mnew = fmaxf(mrow[r], mx);
                    const float corr = __expf(mrow[r] - mnew);
                    mrow[r] = mnew;

                    const float p0 = __expf(v0 - mnew);
                    const float p1 = __expf(v1 - mnew);
                    lrow[r] = lrow[r] * corr + rsum16(p0 + p1);

                    o0[r] *= corr; o1[r] *= corr; o2[r] *= corr; o3[r] *= corr;

                    plds[w][r + hi8][am]      = (_Float16)p0;
                    plds[w][r + hi8][16 + am] = (_Float16)p1;
                }

                // ----- P: C-layout -> A-layout via per-wave LDS slab -----
                v16h pf;
#pragma unroll
                for (int i2 = 0; i2 < 16; ++i2) pf[i2] = plds[w][am][ka_idx(i2, lane)];

                // ----- O += P(16x32) x V(32x64) -----
                v16h vf0, vf1, vf2, vf3;
#pragma unroll
                for (int i2 = 0; i2 < 16; ++i2) {
                    vf0[i2] = vt_tile[cb][0  + am][i2 + hi16];
                    vf1[i2] = vt_tile[cb][16 + am][i2 + hi16];
                    vf2[i2] = vt_tile[cb][32 + am][i2 + hi16];
                    vf3[i2] = vt_tile[cb][48 + am][i2 + hi16];
                }
                o0 = wmma_f16(pf, vf0, o0);
                o1 = wmma_f16(pf, vf1, o1);
                o2 = wmma_f16(pf, vf2, o2);
                o3 = wmma_f16(pf, vf3, o3);
            }

            // transpose-store next V tile (loadcnt wait lands after the WMMAs)
            if (have_next) {
#pragma unroll
                for (int j = 0; j < 8; ++j) vt_tile[(i + 1) & 1][sd + j][skey] = vv[j];
            }
        }

        // ----- normalize and write y[b, t, h*64 + d] (f16) -----
#pragma unroll
        for (int r = 0; r < 8; ++r) {
            const float inv = 1.0f / lrow[r];
            const size_t orow = ((size_t)b * T + qbase + r + hi8) * Cc + (size_t)h * 64;
            y[orow + 0  + am] = (_Float16)(o0[r] * inv);
            y[orow + 16 + am] = (_Float16)(o1[r] * inv);
            y[orow + 32 + am] = (_Float16)(o2[r] * inv);
            y[orow + 48 + am] = (_Float16)(o3[r] * inv);
        }
    }
}

// ---------------------------------------------------------------------------
// Host launch: cvt(x,w_attn,w_proj) -> gemm(qkv) -> flash -> gemm(proj)
// ---------------------------------------------------------------------------
extern "C" void kernel_launch(void* const* d_in, const int* in_sizes, int n_in,
                              void* d_out, int out_size, void* d_ws, size_t ws_size,
                              hipStream_t stream) {
    const float* x      = (const float*)d_in[0];   // [2,4096,512]
    const float* w_attn = (const float*)d_in[1];   // [1536,512]
    const float* b_attn = (const float*)d_in[2];   // [1536]
    const float* w_proj = (const float*)d_in[3];   // [512,512]
    const float* b_proj = (const float*)d_in[4];   // [512]
    float* out = (float*)d_out;                    // [2,4096,512] f32

    constexpr int B = 2, T = 4096, Cc = 512, C3 = 1536;
    constexpr int M = B * T;                       // 8192

    char* ws = (char*)d_ws;
    size_t off = 0;
    _Float16* qkv_h = (_Float16*)(ws + off); off += (size_t)M * C3 * 2;   // 25.2 MB
    _Float16* y_h   = (_Float16*)(ws + off); off += (size_t)M * Cc * 2;   //  8.4 MB
    _Float16* x_h   = (_Float16*)(ws + off); off += (size_t)M * Cc * 2;   //  8.4 MB
    _Float16* wa_h  = (_Float16*)(ws + off); off += (size_t)C3 * Cc * 2;  //  1.6 MB
    _Float16* wp_h  = (_Float16*)(ws + off); off += (size_t)Cc * Cc * 2;  //  0.5 MB

    // 0) one-time f32->f16 conversions (bandwidth-trivial at 23.3 TB/s)
    cvt16<<<(M * Cc / 4 + 255) / 256, 256, 0, stream>>>(x, x_h, M * Cc / 4);
    cvt16<<<(C3 * Cc / 4 + 255) / 256, 256, 0, stream>>>(w_attn, wa_h, C3 * Cc / 4);
    cvt16<<<(Cc * Cc / 4 + 255) / 256, 256, 0, stream>>>(w_proj, wp_h, Cc * Cc / 4);

    // 1) QKV projection: (8192/16) x (1536/128) = 6144 blocks
    gemm16lds<_Float16><<<(M / 16) * (C3 / 128), 256, 0, stream>>>(
        x_h, wa_h, b_attn, qkv_h, M, C3, Cc);

    // 2) Flash attention: B*H*(32/2 stripe-pairs) = 256 uniform-work blocks
    flash16<<<B * 8 * (T / 128 / 2), 256, 0, stream>>>(qkv_h, y_h);

    // 3) Output projection: (8192/16) x (512/128) = 2048 blocks
    gemm16lds<float><<<(M / 16) * (Cc / 128), 256, 0, stream>>>(
        y_h, wp_h, b_proj, out, M, Cc, Cc);
}
